// CustomGPT2Layer_47957604827295
// MI455X (gfx1250) — matrix operand order repaired and checked by reading, not compiled, not run on hardware
//
#include <hip/hip_runtime.h>
#include <hip/hip_bf16.h>

// ---------------------------------------------------------------- constants
#define B_ 4
#define T_ 2048
#define C_ 1024
#define H_ 16
#define D_ 64

typedef __attribute__((ext_vector_type(16))) _Float16 v16h;
typedef __attribute__((ext_vector_type(8)))  float    v8f;

union AFrag { v16h v; uint4 q[2]; };
union BFrag { v16h v; uint4 q[2]; _Float16 h[16]; };

// ================================================================ LayerNorm
// one block per row; 256 threads; f32 in -> f16 out
__global__ __launch_bounds__(256)
void ln_kernel(const float* __restrict__ x, const float* __restrict__ g,
               const float* __restrict__ b, _Float16* __restrict__ out)
{
    const int row = blockIdx.x;
    const int tid = threadIdx.x;
    const float4 v = ((const float4*)(x + (size_t)row * C_))[tid];

    float s  = v.x + v.y + v.z + v.w;
    float s2 = v.x*v.x + v.y*v.y + v.z*v.z + v.w*v.w;
#pragma unroll
    for (int m = 16; m >= 1; m >>= 1) {
        s  += __shfl_xor(s,  m, 32);
        s2 += __shfl_xor(s2, m, 32);
    }
    __shared__ float rs[8], rs2[8];
    if ((tid & 31) == 0) { rs[tid >> 5] = s; rs2[tid >> 5] = s2; }
    __syncthreads();
    s = 0.f; s2 = 0.f;
#pragma unroll
    for (int i = 0; i < 8; ++i) { s += rs[i]; s2 += rs2[i]; }

    const float mu  = s * (1.0f / C_);
    const float var = s2 * (1.0f / C_) - mu * mu;
    const float inv = rsqrtf(var + 1e-5f);

    const int c = tid * 4;
    _Float16* o = out + (size_t)row * C_ + c;
    o[0] = (_Float16)((v.x - mu) * inv * g[c + 0] + b[c + 0]);
    o[1] = (_Float16)((v.y - mu) * inv * g[c + 1] + b[c + 1]);
    o[2] = (_Float16)((v.z - mu) * inv * g[c + 2] + b[c + 2]);
    o[3] = (_Float16)((v.w - mu) * inv * g[c + 3] + b[c + 3]);
}

// ================================================================ GEMM
// C[M,N] = A[M,K](f16) * W[K,N](f32, converted to f16 in LDS) + bias + epilogue
// block tile 128x128, K-step 32, 8 waves (4x2), wave tile 32x64 (2x4 WMMA)
enum { EPI_F16 = 0, EPI_RES_F32 = 1, EPI_GELU_F16 = 2 };

#define LDA 32
#define LDB 40   // 80 bytes/row: every fragment read stays 16B-aligned

template <int EPI>
__global__ __launch_bounds__(256)
void gemm_wmma(const _Float16* __restrict__ A, const float* __restrict__ W,
               const float* __restrict__ bias, const float* __restrict__ resid,
               float* __restrict__ outF, _Float16* __restrict__ outH,
               int M, int N, int K)
{
    __shared__ __align__(16) _Float16 sA[128 * LDA];
    __shared__ __align__(16) _Float16 sB[128 * LDB];

    const int tid  = threadIdx.x;
    const int lane = tid & 31;
    const int wave = tid >> 5;
    const int wm   = wave & 3;   // 4 wave-rows
    const int wn   = wave >> 2;  // 2 wave-cols
    const int bm   = blockIdx.y * 128;
    const int bn   = blockIdx.x * 128;
    const int ln   = lane & 15;
    const int hi   = lane >> 4;

    v8f acc[2][4];
#pragma unroll
    for (int i = 0; i < 2; ++i)
#pragma unroll
        for (int j = 0; j < 4; ++j)
#pragma unroll
            for (int e = 0; e < 8; ++e) acc[i][j][e] = 0.f;

    for (int k0 = 0; k0 < K; k0 += 32) {
        // --- stage A (f16, row-major [M][K]) : 128x32 halves
        {
            const int row = tid >> 1;
            const int seg = (tid & 1) * 16;
            const uint4* gp = (const uint4*)(A + (size_t)(bm + row) * K + k0 + seg);
            uint4* sp = (uint4*)(sA + row * LDA + seg);
            sp[0] = gp[0];
            sp[1] = gp[1];
        }
        // --- stage B transposed: W[K][N] f32 -> sB[n][k] f16
#pragma unroll
        for (int i = 0; i < 4; ++i) {
            const int r = i * 8 + (tid >> 5);    // k-local 0..31
            const int c = (tid & 31) * 4;        // n-local
            const float* wp = W + (size_t)(k0 + r) * N + bn + c;
            const float4 w = *(const float4*)wp;
            // prefetch next K-tile of the f32 weight stream (global_prefetch_b8)
            if (k0 + 32 < K) __builtin_prefetch(wp + (size_t)32 * N, 0, 1);
            sB[(c + 0) * LDB + r] = (_Float16)w.x;
            sB[(c + 1) * LDB + r] = (_Float16)w.y;
            sB[(c + 2) * LDB + r] = (_Float16)w.z;
            sB[(c + 3) * LDB + r] = (_Float16)w.w;
        }
        __syncthreads();

        AFrag a[2];
        BFrag bfr[4];
#pragma unroll
        for (int t = 0; t < 2; ++t) {
            const int m = wm * 32 + t * 16 + ln;
            const uint4* p = (const uint4*)(sA + m * LDA);
            a[t].q[0] = p[2 * hi];       // K = hi*16 .. +7
            a[t].q[1] = p[2 * hi + 1];   // K = hi*16+8 .. +7
        }
#pragma unroll
        for (int t = 0; t < 4; ++t) {
            const int n = wn * 64 + t * 16 + ln;
            const uint4* p = (const uint4*)(sB + n * LDB + hi * 16);
            bfr[t].q[0] = p[0];
            bfr[t].q[1] = p[1];
        }
#pragma unroll
        for (int i = 0; i < 2; ++i)
#pragma unroll
            for (int j = 0; j < 4; ++j)
                acc[i][j] = __builtin_amdgcn_wmma_f32_16x16x32_f16(
                    false, a[i].v, false, bfr[j].v, (short)0, acc[i][j], false, false);
        __syncthreads();
    }

    // --- epilogue
#pragma unroll
    for (int i = 0; i < 2; ++i)
#pragma unroll
        for (int j = 0; j < 4; ++j) {
            const int col = bn + wn * 64 + j * 16 + ln;
            const float bv = bias[col];
#pragma unroll
            for (int r = 0; r < 8; ++r) {
                const int row = bm + wm * 32 + i * 16 + r + hi * 8;
                const size_t idx = (size_t)row * N + col;
                float v = acc[i][j][r] + bv;
                if (EPI == EPI_F16) {
                    outH[idx] = (_Float16)v;
                } else if (EPI == EPI_RES_F32) {
                    outF[idx] = v + resid[idx];
                } else { // exact-erf GELU
                    outH[idx] = (_Float16)(0.5f * v * (1.0f + erff(v * 0.70710678118f)));
                }
            }
        }
}

// ================================================================ attention
// flash-style: one wave per (b,h, 16-query tile). Single-wave workgroup so
// __syncthreads is free (S_NOP) but fences LDS for the compiler.
__global__ __launch_bounds__(32)
void attn_flash(const _Float16* __restrict__ qkv, _Float16* __restrict__ y)
{
    __shared__ __align__(16) _Float16 sP[16 * 32];   // P tile, row-major [q][key]
    __shared__ __align__(16) _Float16 sV[32 * 72];   // V tile, row-major [key][d]

    const int lane = threadIdx.x;
    const int ln = lane & 15, hi = lane >> 4;
    const int q0 = blockIdx.x * 16;
    const int bh = blockIdx.y;
    const int b  = bh >> 4;
    const int h  = bh & 15;

    const size_t rstride = 3 * C_;
    const _Float16* qbase = qkv + (size_t)b * T_ * rstride + h * D_;
    const _Float16* kbase = qbase + C_;
    const _Float16* vbase = qbase + 2 * C_;

    // Q A-fragments (reused for every key tile): rows q0+ln, K-dim = d
    AFrag aq[2];
    {
        const _Float16* qr = qbase + (size_t)(q0 + ln) * rstride;
#pragma unroll
        for (int d = 0; d < 2; ++d) {
            aq[d].q[0] = *(const uint4*)(qr + d * 32 + hi * 16);
            aq[d].q[1] = *(const uint4*)(qr + d * 32 + hi * 16 + 8);
        }
    }

    float mrun[8], lrun[8];
    v8f o[4];
#pragma unroll
    for (int r = 0; r < 8; ++r) { mrun[r] = -1e30f; lrun[r] = 0.f; }
#pragma unroll
    for (int c = 0; c < 4; ++c)
#pragma unroll
        for (int e = 0; e < 8; ++e) o[c][e] = 0.f;

    const float scale = 0.125f; // 1/sqrt(64)
    const int nk = q0 + 16;     // causal: keys 0..q0+15

    for (int kb = 0; kb < nk; kb += 32) {
        // stage V[kb..kb+31][0..31] into LDS (coalesced rows)
        {
            const uint4* src = (const uint4*)(vbase + (size_t)(kb + lane) * rstride);
            uint4* dst = (uint4*)(sV + lane * 72);
            dst[0] = src[0]; dst[1] = src[1]; dst[2] = src[2]; dst[3] = src[3];
        }

        // S = Q * K^T for two 16-key tiles; K rows are d-contiguous => direct B frags
        v8f s[2];
#pragma unroll
        for (int st = 0; st < 2; ++st) {
#pragma unroll
            for (int e = 0; e < 8; ++e) s[st][e] = 0.f;
            const int key = kb + st * 16 + ln;
            const _Float16* kr = kbase + (size_t)key * rstride;
#pragma unroll
            for (int d = 0; d < 2; ++d) {
                BFrag bk;
                bk.q[0] = *(const uint4*)(kr + d * 32 + hi * 16);
                bk.q[1] = *(const uint4*)(kr + d * 32 + hi * 16 + 8);
                s[st] = __builtin_amdgcn_wmma_f32_16x16x32_f16(
                    false, aq[d].v, false, bk.v, (short)0, s[st], false, false);
            }
        }
        __syncthreads();

        // online softmax over this 32-key chunk (row reductions within 16-lane groups)
#pragma unroll
        for (int r = 0; r < 8; ++r) {
            const int row = q0 + r + hi * 8;
            float v0 = s[0][r] * scale;
            float v1 = s[1][r] * scale;
            if (kb + ln > row)      v0 = -1e30f;
            if (kb + 16 + ln > row) v1 = -1e30f;
            float mx = fmaxf(v0, v1);
#pragma unroll
            for (int msk = 8; msk >= 1; msk >>= 1) mx = fmaxf(mx, __shfl_xor(mx, msk, 32));
            const float mnew = fmaxf(mrun[r], mx);
            const float f  = __expf(mrun[r] - mnew);
            const float p0 = __expf(v0 - mnew);
            const float p1 = __expf(v1 - mnew);
            float ls = p0 + p1;
#pragma unroll
            for (int msk = 8; msk >= 1; msk >>= 1) ls += __shfl_xor(ls, msk, 32);
            lrun[r] = lrun[r] * f + ls;
            mrun[r] = mnew;
#pragma unroll
            for (int c = 0; c < 4; ++c) o[c][r] *= f;
            const int rl = r + hi * 8;               // C-layout -> row-major P in LDS
            sP[rl * 32 + ln]      = (_Float16)p0;
            sP[rl * 32 + 16 + ln] = (_Float16)p1;
        }
        __syncthreads();

        // O += P * V
        AFrag ap;
        {
            const uint4* p = (const uint4*)(sP + ln * 32);
            ap.q[0] = p[2 * hi];
            ap.q[1] = p[2 * hi + 1];
        }
#pragma unroll
        for (int c = 0; c < 4; ++c) {
            BFrag bv;
#pragma unroll
            for (int e = 0; e < 16; ++e)
                bv.h[e] = sV[(hi * 16 + e) * 72 + c * 16 + ln];
            o[c] = __builtin_amdgcn_wmma_f32_16x16x32_f16(
                false, ap.v, false, bv.v, (short)0, o[c], false, false);
        }
        __syncthreads();
    }

    // normalize + store y[b][t][h*D + d]
#pragma unroll
    for (int c = 0; c < 4; ++c)
#pragma unroll
        for (int r = 0; r < 8; ++r) {
            const int t = q0 + r + hi * 8;
            const float val = o[c][r] / lrun[r];
            y[((size_t)b * T_ + t) * C_ + h * D_ + c * 16 + ln] = (_Float16)val;
        }
}

// ================================================================ launcher
extern "C" void kernel_launch(void* const* d_in, const int* in_sizes, int n_in,
                              void* d_out, int out_size, void* d_ws, size_t ws_size,
                              hipStream_t stream)
{
    const float* x      = (const float*)d_in[0];
    const float* w_attn = (const float*)d_in[1];
    const float* b_attn = (const float*)d_in[2];
    const float* w_proj = (const float*)d_in[3];
    const float* b_proj = (const float*)d_in[4];
    const float* ln1_g  = (const float*)d_in[5];
    const float* ln1_b  = (const float*)d_in[6];
    const float* ln2_g  = (const float*)d_in[7];
    const float* ln2_b  = (const float*)d_in[8];
    const float* w_fc   = (const float*)d_in[9];
    const float* b_fc   = (const float*)d_in[10];
    const float* w_fc2  = (const float*)d_in[11];
    const float* b_fc2  = (const float*)d_in[12];

    char* ws = (char*)d_ws;
    const size_t MB = 1024u * 1024u;
    _Float16* ln_buf = (_Float16*)(ws);              // 16 MB  [8192,1024] f16
    _Float16* qkvb   = (_Float16*)(ws + 16 * MB);    // 48 MB  [8192,3072] f16
    _Float16* ybuf   = (_Float16*)(ws + 64 * MB);    // 16 MB  [8192,1024] f16
    _Float16* fc_h   = (_Float16*)(ws + 16 * MB);    // 64 MB  [8192,4096] f16 (reuses qkv+y)
    float*    resid2 = (float*)   (ws + 80 * MB);    // 32 MB  [8192,1024] f32

    const int M = B_ * T_; // 8192

    // 1) LN1 -> f16
    ln_kernel<<<M, 256, 0, stream>>>(x, ln1_g, ln1_b, ln_buf);

    // 2) QKV = ln1 @ w_attn + b_attn   [8192,3072] f16
    gemm_wmma<EPI_F16><<<dim3(3072 / 128, M / 128), 256, 0, stream>>>(
        ln_buf, w_attn, b_attn, nullptr, nullptr, qkvb, M, 3072, C_);

    // 3) causal flash attention -> y f16
    attn_flash<<<dim3(T_ / 16, B_ * H_), 32, 0, stream>>>(qkvb, ybuf);

    // 4) resid2 = y @ w_proj + b_proj + x   [8192,1024] f32
    gemm_wmma<EPI_RES_F32><<<dim3(C_ / 128, M / 128), 256, 0, stream>>>(
        ybuf, w_proj, b_proj, x, resid2, nullptr, M, C_, C_);

    // 5) LN2 -> f16
    ln_kernel<<<M, 256, 0, stream>>>(resid2, ln2_g, ln2_b, ln_buf);

    // 6) fc_h = gelu(ln2 @ w_fc + b_fc)   [8192,4096] f16
    gemm_wmma<EPI_GELU_F16><<<dim3(4096 / 128, M / 128), 256, 0, stream>>>(
        ln_buf, w_fc, b_fc, nullptr, nullptr, fc_h, M, 4096, C_);

    // 7) out = fc_h @ w_fc2 + b_fc2 + resid2   [8192,1024] f32
    gemm_wmma<EPI_RES_F32><<<dim3(C_ / 128, M / 128), 256, 0, stream>>>(
        fc_h, w_fc2, b_fc2, resid2, (float*)d_out, nullptr, M, C_, 4096);
}
